// MoDBlock_65687229825689
// MI455X (gfx1250) — matrix operand order repaired and checked by reading, not compiled, not run on hardware
//
#include <hip/hip_runtime.h>

// ---------------------------------------------------------------------------
// MoD transformer block for MI455X (gfx1250), wave32 + WMMA bf16 + TDM.
// B=4 T=4096 D=1024 H=16 hd=64 DFF=4096 capacity=0.5 -> k=2048, M=B*k=8192.
// ~340 GFLOP of GEMM vs ~300MB traffic (13us @ 23.3TB/s) => matrix-engine
// bound => every contraction through V_WMMA_F32_16X16X32_BF16, and the main
// GEMM K-loop staged by the Tensor Data Mover (double-buffered, TENSORcnt).
// ---------------------------------------------------------------------------

#define DMODEL 1024
#define NHEAD  16
#define HD     64
#define TSEQ   4096
#define BATCH  4
#define DFFDIM 4096
#define KSEL   2048
#define EPSLN  1e-5f

typedef __bf16 v16bf __attribute__((ext_vector_type(16)));
typedef float  v8f   __attribute__((ext_vector_type(8)));
typedef unsigned int u32x4 __attribute__((ext_vector_type(4)));
typedef int          i32x4 __attribute__((ext_vector_type(4)));
typedef int          i32x8 __attribute__((ext_vector_type(8)));

union FragB16 {          // one WMMA 16x16x32 bf16 operand (8 VGPRs / lane)
    v16bf v;
    uint4 q[2];
    unsigned short s[16];
};

__device__ __forceinline__ unsigned short f2bf(float f) {
    unsigned u = __builtin_bit_cast(unsigned, f);
    u += 0x7FFFu + ((u >> 16) & 1u);          // round-to-nearest-even
    return (unsigned short)(u >> 16);
}

__device__ __forceinline__ v8f v8f_zero() {
    v8f z;
#pragma unroll
    for (int i = 0; i < 8; ++i) z[i] = 0.0f;
    return z;
}

__device__ __forceinline__ v8f wmma_bf16(const FragB16& a, const FragB16& b, v8f c) {
    return __builtin_amdgcn_wmma_f32_16x16x32_bf16(
        /*neg_a=*/false, a.v, /*neg_b=*/false, b.v,
        /*c_mod=*/(short)0, c, /*reuse_a=*/false, /*reuse_b=*/false);
}

// ---------------------------------------------------------------------------
// TDM 2-D tile load: global (rows x cols bf16, row stride in elements) -> LDS.
// D# per CDNA5 ISA ch.8: group0 = {count, lds_addr, global_addr, type=2},
// group1 = {data_size=2B, tensor dims, tile dims, dim0 stride}.  Issued by a
// single wave; completion tracked with TENSORcnt (s_wait_tensorcnt).
// ---------------------------------------------------------------------------
__device__ __forceinline__ void tdm_load_2d(const void* gsrc, unsigned lds_off,
                                            int rows, int cols, int row_stride) {
    unsigned long long ga = (unsigned long long)gsrc;
    u32x4 g0;
    g0[0] = 1u;                                            // count=1 valid D#
    g0[1] = lds_off;                                       // lds_addr (bytes)
    g0[2] = (unsigned)(ga & 0xFFFFFFFFull);                // global_addr[31:0]
    g0[3] = (unsigned)((ga >> 32) & 0x1FFFFFFull)          // global_addr[56:32]
          | (2u << 30);                                    // type = 2 ("image")
    i32x8 g1;
    g1[0] = 1 << 16;                                       // data_size=1 -> 2B
    g1[1] = (cols & 0xFFFF) << 16;                         // tensor_dim0[15:0]
    g1[2] = ((unsigned)cols >> 16) | ((rows & 0xFFFF) << 16); // dim0 hi, dim1 lo
    g1[3] = ((unsigned)rows >> 16) | ((cols & 0xFFFF) << 16); // dim1 hi, tile_dim0
    g1[4] = rows & 0xFFFF;                                 // tile_dim1 (dim2=0)
    g1[5] = row_stride;                                    // dim0_stride[31:0]
    g1[6] = 0;
    g1[7] = 0;
    i32x4 z4 = {0, 0, 0, 0};
#if __clang_major__ >= 23
    i32x8 z8 = {0, 0, 0, 0, 0, 0, 0, 0};
    __builtin_amdgcn_tensor_load_to_lds(g0, g1, z4, z4, z8, 0);
#else
    __builtin_amdgcn_tensor_load_to_lds(g0, g1, z4, z4, 0);
#endif
}

// ---------------------------------------------------------------------------
// fp32 -> bf16 weight conversion (one-time; weights then live in L2, 192MB)
// ---------------------------------------------------------------------------
__global__ __launch_bounds__(256) void cvt_bf16_kernel(
    const float* __restrict__ in, unsigned short* __restrict__ out, int n) {
    int i = blockIdx.x * 256 + threadIdx.x;
    if (i < n) out[i] = f2bf(in[i]);
}

// ---------------------------------------------------------------------------
// Router: score[b,t] = sigmoid(dot(x[b,t,:], w_router)).  One wave per token.
// ---------------------------------------------------------------------------
__global__ __launch_bounds__(256) void router_kernel(
    const float* __restrict__ x, const float* __restrict__ wr,
    float* __restrict__ scores) {
    int wave = threadIdx.x >> 5, lane = threadIdx.x & 31;
    int row  = blockIdx.x * 8 + wave;            // < B*T
    const float* xr = x + (size_t)row * DMODEL;
    float s = 0.f;
#pragma unroll
    for (int j = 0; j < DMODEL / 32; ++j) {
        int d = j * 32 + lane;
        s += xr[d] * wr[d];
    }
#pragma unroll
    for (int off = 16; off > 0; off >>= 1) s += __shfl_xor(s, off, 32);
    if (lane == 0) scores[row] = 1.f / (1.f + __expf(-s));
}

// ---------------------------------------------------------------------------
// Top-k per batch (k = T/2).  Bitonic sort of (score,idx) keys in LDS,
// descending score with lower-index-first tiebreak, then ascending index
// sort of the selected k (matches jnp.sort(top_k idx)).
// ---------------------------------------------------------------------------
__global__ __launch_bounds__(1024) void topk_kernel(
    const float* __restrict__ scores, int* __restrict__ idxout) {
    __shared__ unsigned long long keys[TSEQ];   // 32 KB
    __shared__ unsigned int       sel[KSEL];    //  8 KB
    int b = blockIdx.x, tid = threadIdx.x;
    for (int i = tid; i < TSEQ; i += 1024) {
        unsigned u = __builtin_bit_cast(unsigned, scores[b * TSEQ + i]);
        u = (u & 0x80000000u) ? ~u : (u | 0x80000000u);   // order-preserving map
        keys[i] = ((unsigned long long)u << 32) |
                  (unsigned long long)(0xFFFFFFFFu - (unsigned)i);
    }
    __syncthreads();
    for (int size = 2; size <= TSEQ; size <<= 1)
        for (int stride = size >> 1; stride > 0; stride >>= 1) {
            for (int i = tid; i < TSEQ; i += 1024) {
                int j = i ^ stride;
                if (j > i) {
                    bool up = ((i & size) == 0);
                    unsigned long long a = keys[i], c = keys[j];
                    if ((a < c) == up) { keys[i] = c; keys[j] = a; }  // descending
                }
            }
            __syncthreads();
        }
    for (int i = tid; i < KSEL; i += 1024)
        sel[i] = 0xFFFFFFFFu - (unsigned)(keys[i] & 0xFFFFFFFFull);
    __syncthreads();
    for (int size = 2; size <= KSEL; size <<= 1)
        for (int stride = size >> 1; stride > 0; stride >>= 1) {
            for (int i = tid; i < KSEL; i += 1024) {
                int j = i ^ stride;
                if (j > i) {
                    bool up = ((i & size) == 0);
                    unsigned a = sel[i], c = sel[j];
                    if ((a > c) == up) { sel[i] = c; sel[j] = a; }    // ascending
                }
            }
            __syncthreads();
        }
    for (int i = tid; i < KSEL; i += 1024) idxout[b * KSEL + i] = (int)sel[i];
}

// ---------------------------------------------------------------------------
// aux_loss = var(mean_t scores, ddof=1).  One wave per batch.
// ---------------------------------------------------------------------------
__global__ __launch_bounds__(128) void aux_kernel(
    const float* __restrict__ scores, float* __restrict__ out_aux) {
    __shared__ float mb[BATCH];
    int wave = threadIdx.x >> 5, lane = threadIdx.x & 31;
    float s = 0.f;
    for (int i = lane; i < TSEQ; i += 32) s += scores[wave * TSEQ + i];
#pragma unroll
    for (int off = 16; off > 0; off >>= 1) s += __shfl_xor(s, off, 32);
    if (lane == 0) mb[wave] = s / (float)TSEQ;
    __syncthreads();
    if (threadIdx.x == 0) {
        float mu = 0.25f * (mb[0] + mb[1] + mb[2] + mb[3]);
        float v = 0.f;
        for (int i = 0; i < BATCH; ++i) { float d = mb[i] - mu; v += d * d; }
        out_aux[0] = v / (float)(BATCH - 1);
    }
}

// ---------------------------------------------------------------------------
// LayerNorm (optionally gathering rows through idx), fp32 in -> bf16 out.
// One wave per token row; values kept in registers for the second pass.
// ---------------------------------------------------------------------------
__global__ __launch_bounds__(256) void ln_kernel(
    const float* __restrict__ src, const int* __restrict__ idx,
    const float* __restrict__ scale, const float* __restrict__ bias,
    unsigned short* __restrict__ out, int useGather) {
    int wave = threadIdx.x >> 5, lane = threadIdx.x & 31;
    int row  = blockIdx.x * 8 + wave;            // < B*KSEL
    size_t srow;
    if (useGather) {
        int b = row / KSEL, r = row % KSEL;
        srow = (size_t)b * TSEQ + (size_t)idx[b * KSEL + r];
    } else {
        srow = (size_t)row;
    }
    const float* xp = src + srow * DMODEL;
    float vals[32];
    float s = 0.f, s2 = 0.f;
#pragma unroll
    for (int j = 0; j < 32; ++j) {
        float v = xp[j * 32 + lane];
        vals[j] = v; s += v; s2 += v * v;
    }
#pragma unroll
    for (int off = 16; off > 0; off >>= 1) {
        s  += __shfl_xor(s,  off, 32);
        s2 += __shfl_xor(s2, off, 32);
    }
    float mean = s * (1.0f / DMODEL);
    float var  = s2 * (1.0f / DMODEL) - mean * mean;
    float rstd = rsqrtf(var + EPSLN);
    unsigned short* op = out + (size_t)row * DMODEL;
#pragma unroll
    for (int j = 0; j < 32; ++j) {
        int d = j * 32 + lane;
        op[d] = f2bf((vals[j] - mean) * rstd * scale[d] + bias[d]);
    }
}

// ---------------------------------------------------------------------------
// Tiled WMMA GEMM with double-buffered TDM staging:
//   C[m,n] = sum_k A[m,k] * W[n,k]   (A,W bf16, C f32 acc)
// Block tile 128x128, 8 waves in a 4x2 grid, 32x64 (2x4 WMMA tiles) / wave.
// Per K-step: entry barrier (all reads of the alternate buffer done) ->
// wave 0 issues TDM for tile i+1 -> s_wait_tensorcnt 2 (tile i complete,
// tile i+1 still in flight) -> barrier -> WMMA compute overlaps the DMA.
// EPI 0: store bf16.   EPI 1: +gather(x) residual, store f32 (out-proj).
// EPI 2: +res, scatter into (B,T,D) output through idx (final FFN).
// ---------------------------------------------------------------------------
template <int EPI>
__global__ __launch_bounds__(256) void gemm_kernel(
    const unsigned short* __restrict__ A, const unsigned short* __restrict__ W,
    int M, int N, int K, void* __restrict__ outp,
    const float* __restrict__ gatherX, const float* __restrict__ res,
    const int* __restrict__ idx) {
    __shared__ __align__(16) unsigned short sA[2][128 * 32];
    __shared__ __align__(16) unsigned short sW[2][128 * 32];
    int tid = threadIdx.x, wave = tid >> 5, lane = tid & 31;
    int bm = blockIdx.y * 128, bn = blockIdx.x * 128;
    int wm = (wave & 3) * 32, wn = (wave >> 2) * 64;
    int g = lane >> 4, lm = lane & 15;

    v8f acc[2][4];
#pragma unroll
    for (int ti = 0; ti < 2; ++ti)
#pragma unroll
        for (int tj = 0; tj < 4; ++tj) acc[ti][tj] = v8f_zero();

    if (wave == 0) {   // prologue: DMA tile 0 into buffer 0
        tdm_load_2d(A + (size_t)bm * K, (unsigned)(size_t)&sA[0][0], 128, 32, K);
        tdm_load_2d(W + (size_t)bn * K, (unsigned)(size_t)&sW[0][0], 128, 32, K);
    }

    int buf = 0;
    for (int kk = 0; kk < K; kk += 32, buf ^= 1) {
        __syncthreads();               // alternate buffer free for DMA reuse
        if (wave == 0) {
            if (kk + 32 < K) {
                tdm_load_2d(A + (size_t)bm * K + kk + 32,
                            (unsigned)(size_t)&sA[buf ^ 1][0], 128, 32, K);
                tdm_load_2d(W + (size_t)bn * K + kk + 32,
                            (unsigned)(size_t)&sW[buf ^ 1][0], 128, 32, K);
                __builtin_amdgcn_s_wait_tensorcnt(2);   // tile kk landed
            } else {
                __builtin_amdgcn_s_wait_tensorcnt(0);
            }
        }
        __syncthreads();               // tile kk visible to all waves

        FragB16 af[2], bf2[4];
#pragma unroll
        for (int t = 0; t < 2; ++t) {
            // A fragment layout (ISA 7.12.2, 16-bit A 16x32)
            const unsigned short* pa = &sA[buf][(wm + t * 16 + lm) * 32];
            af[t].q[0] = *(const uint4*)(pa + 8 * g);
            af[t].q[1] = *(const uint4*)(pa + 16 + 8 * g);
        }
#pragma unroll
        for (int t = 0; t < 4; ++t) {
            // B fragment layout (same K order as A: 8*half, 16+8*half)
            const unsigned short* pb = &sW[buf][(wn + t * 16 + lm) * 32];
            bf2[t].q[0] = *(const uint4*)(pb + 8 * g);
            bf2[t].q[1] = *(const uint4*)(pb + 16 + 8 * g);
        }
#pragma unroll
        for (int ti = 0; ti < 2; ++ti)
#pragma unroll
            for (int tj = 0; tj < 4; ++tj)
                acc[ti][tj] = wmma_bf16(af[ti], bf2[tj], acc[ti][tj]);
    }

#pragma unroll
    for (int ti = 0; ti < 2; ++ti)
#pragma unroll
        for (int tj = 0; tj < 4; ++tj)
#pragma unroll
            for (int r = 0; r < 8; ++r) {
                int m = bm + wm + ti * 16 + r + 8 * g;     // C layout: row r+8*half
                int n = bn + wn + tj * 16 + lm;            //           col lane&15
                float v = acc[ti][tj][r];
                if (EPI == 0) {
                    ((unsigned short*)outp)[(size_t)m * N + n] = f2bf(v);
                } else if (EPI == 1) {
                    int b = m / KSEL, rr = m % KSEL;
                    int t = idx[b * KSEL + rr];
                    ((float*)outp)[(size_t)m * N + n] =
                        v + gatherX[((size_t)b * TSEQ + t) * DMODEL + n];
                } else {
                    int b = m / KSEL, rr = m % KSEL;
                    int t = idx[b * KSEL + rr];
                    ((float*)outp)[((size_t)b * TSEQ + t) * DMODEL + n] =
                        v + res[(size_t)m * N + n];
                }
            }
}

// ---------------------------------------------------------------------------
// Fused SwiGLU dual-GEMM: g = silu(A@W1^T) * (A@W2^T), bf16 out.
// Vector-load staging (compiler software-pipelines global->LDS->WMMA).
// ---------------------------------------------------------------------------
__global__ __launch_bounds__(256) void gemm_swiglu_kernel(
    const unsigned short* __restrict__ A, const unsigned short* __restrict__ W1,
    const unsigned short* __restrict__ W2, int M, int N, int K,
    unsigned short* __restrict__ outp) {
    __shared__ __align__(16) unsigned short sA[128 * 32];
    __shared__ __align__(16) unsigned short sW1[64 * 32];
    __shared__ __align__(16) unsigned short sW2[64 * 32];
    int tid = threadIdx.x, wave = tid >> 5, lane = tid & 31;
    int bm = blockIdx.y * 128, bn = blockIdx.x * 64;
    int wm = (wave & 3) * 32, wn = (wave >> 2) * 32;
    int g = lane >> 4, lm = lane & 15;

    v8f acc1[2][2], acc2[2][2];
#pragma unroll
    for (int ti = 0; ti < 2; ++ti)
#pragma unroll
        for (int tj = 0; tj < 2; ++tj) { acc1[ti][tj] = v8f_zero(); acc2[ti][tj] = v8f_zero(); }

    for (int kk = 0; kk < K; kk += 32) {
#pragma unroll
        for (int c = 0; c < 2; ++c) {
            int ch = tid + c * 256;
            int row = ch >> 2, seg = ch & 3;
            *(uint4*)(sA + row * 32 + seg * 8) =
                *(const uint4*)(A + (size_t)(bm + row) * K + kk + seg * 8);
        }
        {
            int row = tid >> 2, seg = tid & 3;
            *(uint4*)(sW1 + row * 32 + seg * 8) =
                *(const uint4*)(W1 + (size_t)(bn + row) * K + kk + seg * 8);
            *(uint4*)(sW2 + row * 32 + seg * 8) =
                *(const uint4*)(W2 + (size_t)(bn + row) * K + kk + seg * 8);
        }
        if (kk + 32 < K)   // gfx1250: lowers to global_prefetch_b8
            __builtin_prefetch(A + (size_t)(bm + (tid >> 1)) * K + kk + 32, 0, 1);
        __syncthreads();

        FragB16 af[2], b1[2], b2[2];
#pragma unroll
        for (int t = 0; t < 2; ++t) {
            const unsigned short* pa = sA + (wm + t * 16 + lm) * 32;
            af[t].q[0] = *(const uint4*)(pa + 8 * g);
            af[t].q[1] = *(const uint4*)(pa + 16 + 8 * g);
            const unsigned short* p1 = sW1 + (wn + t * 16 + lm) * 32;
            b1[t].q[0] = *(const uint4*)(p1 + 8 * g);
            b1[t].q[1] = *(const uint4*)(p1 + 16 + 8 * g);
            const unsigned short* p2 = sW2 + (wn + t * 16 + lm) * 32;
            b2[t].q[0] = *(const uint4*)(p2 + 8 * g);
            b2[t].q[1] = *(const uint4*)(p2 + 16 + 8 * g);
        }
#pragma unroll
        for (int ti = 0; ti < 2; ++ti)
#pragma unroll
            for (int tj = 0; tj < 2; ++tj) {
                acc1[ti][tj] = wmma_bf16(af[ti], b1[tj], acc1[ti][tj]);
                acc2[ti][tj] = wmma_bf16(af[ti], b2[tj], acc2[ti][tj]);
            }
        __syncthreads();
    }

#pragma unroll
    for (int ti = 0; ti < 2; ++ti)
#pragma unroll
        for (int tj = 0; tj < 2; ++tj)
#pragma unroll
            for (int r = 0; r < 8; ++r) {
                int m = bm + wm + ti * 16 + r + 8 * g;
                int n = bn + wn + tj * 16 + lm;
                float a = acc1[ti][tj][r], c = acc2[ti][tj][r];
                float silu = a / (1.0f + __expf(-a));
                outp[(size_t)m * N + n] = f2bf(silu * c);
            }
}

// ---------------------------------------------------------------------------
// Flash attention over the k=2048 selected tokens, per (b, h, 64-query tile).
// 4 waves; each wave owns 16 query rows (S strip 16x64 per key tile).
// Online softmax uses 16-lane shuffle row reductions (C-layout rows live
// across 16 lanes).  P is bounced through wave-private LDS to re-layout it
// as a WMMA A-fragment; V is staged transposed so P@V B-fragments are
// contiguous b128 LDS reads.
// ---------------------------------------------------------------------------
__global__ __launch_bounds__(128) void attn_kernel(
    const unsigned short* __restrict__ qkv,   // (B*KSEL) x 3D bf16
    unsigned short* __restrict__ o) {         // (B*KSEL) x D  bf16
    __shared__ __align__(16) unsigned short sK[64 * 64];
    __shared__ __align__(16) unsigned short sVt[64 * 64];
    __shared__ __align__(16) unsigned short sP[4 * 16 * 64];
    int tid = threadIdx.x, wave = tid >> 5, lane = tid & 31;
    int g = lane >> 4, lm = lane & 15;
    int b = blockIdx.z, h = blockIdx.y, q0 = blockIdx.x * 64;
    const float scale = 0.125f;               // 1/sqrt(64)

    FragB16 aq[2];                            // Q 16x64 in A-fragment form
    {
        const unsigned short* qp =
            qkv + ((size_t)b * KSEL + q0 + wave * 16 + lm) * (3 * DMODEL) + h * HD;
#pragma unroll
        for (int ds = 0; ds < 2; ++ds) {
            aq[ds].q[0] = *(const uint4*)(qp + ds * 32 + 8 * g);
            aq[ds].q[1] = *(const uint4*)(qp + ds * 32 + 16 + 8 * g);
        }
    }

    float mstate[8], lstate[8];
    v8f oacc[4];
#pragma unroll
    for (int r = 0; r < 8; ++r) { mstate[r] = -1e30f; lstate[r] = 0.f; }
#pragma unroll
    for (int f = 0; f < 4; ++f) oacc[f] = v8f_zero();

    for (int kt = 0; kt < KSEL / 64; ++kt) {
        // stage K tile (row-major) and V tile (transposed, d-major)
#pragma unroll
        for (int c = 0; c < 4; ++c) {
            int ch = tid + c * 128;
            int kr = ch >> 3, seg = ch & 7;
            *(uint4*)(sK + kr * 64 + seg * 8) =
                *(const uint4*)(qkv + ((size_t)b * KSEL + kt * 64 + kr) * (3 * DMODEL)
                                + DMODEL + h * HD + seg * 8);
        }
#pragma unroll
        for (int c = 0; c < 4; ++c) {
            int ch = tid + c * 128;
            int kr = ch >> 3, seg = ch & 7;
            union { uint4 u; unsigned short s[8]; } tmp;
            tmp.u = *(const uint4*)(qkv + ((size_t)b * KSEL + kt * 64 + kr) * (3 * DMODEL)
                                    + 2 * DMODEL + h * HD + seg * 8);
#pragma unroll
            for (int j = 0; j < 8; ++j) sVt[(seg * 8 + j) * 64 + kr] = tmp.s[j];
        }
        __syncthreads();

        // S = Q @ K^T  (16 queries x 64 keys per wave)
        v8f sacc[4];
#pragma unroll
        for (int f = 0; f < 4; ++f) {
            sacc[f] = v8f_zero();
#pragma unroll
            for (int ds = 0; ds < 2; ++ds) {
                FragB16 bk;
                const unsigned short* p = sK + (f * 16 + lm) * 64 + ds * 32 + 8 * g;
                bk.q[0] = *(const uint4*)p;
                bk.q[1] = *(const uint4*)(p + 16);
                sacc[f] = wmma_bf16(aq[ds], bk, sacc[f]);
            }
        }

        // online softmax, per query row r (and r+8 in the other half-wave)
#pragma unroll
        for (int r = 0; r < 8; ++r) {
#pragma unroll
            for (int f = 0; f < 4; ++f) sacc[f][r] *= scale;
            float mv = fmaxf(fmaxf(sacc[0][r], sacc[1][r]),
                             fmaxf(sacc[2][r], sacc[3][r]));
#pragma unroll
            for (int off = 1; off < 16; off <<= 1) mv = fmaxf(mv, __shfl_xor(mv, off, 32));
            float mnew = fmaxf(mstate[r], mv);
            float corr = __expf(mstate[r] - mnew);
            float rs = 0.f;
#pragma unroll
            for (int f = 0; f < 4; ++f) {
                float p = __expf(sacc[f][r] - mnew);
                sacc[f][r] = p; rs += p;
            }
#pragma unroll
            for (int off = 1; off < 16; off <<= 1) rs += __shfl_xor(rs, off, 32);
            mstate[r] = mnew;
            lstate[r] = lstate[r] * corr + rs;
#pragma unroll
            for (int f = 0; f < 4; ++f) oacc[f][r] *= corr;
        }

        // re-layout P (C-layout f32) -> A-fragment bf16 via wave-private LDS
        unsigned short* pp = sP + wave * 16 * 64;
#pragma unroll
        for (int f = 0; f < 4; ++f)
#pragma unroll
            for (int r = 0; r < 8; ++r)
                pp[(r + 8 * g) * 64 + f * 16 + lm] = f2bf(sacc[f][r]);

        FragB16 ap[2];
#pragma unroll
        for (int ks = 0; ks < 2; ++ks) {
            const unsigned short* p = pp + lm * 64 + ks * 32;
            ap[ks].q[0] = *(const uint4*)(p + 8 * g);
            ap[ks].q[1] = *(const uint4*)(p + 16 + 8 * g);
        }

        // O += P @ V
#pragma unroll
        for (int f = 0; f < 4; ++f)
#pragma unroll
            for (int ks = 0; ks < 2; ++ks) {
                FragB16 bv;
                const unsigned short* p = sVt + (f * 16 + lm) * 64 + ks * 32 + 8 * g;
                bv.q[0] = *(const uint4*)p;
                bv.q[1] = *(const uint4*)(p + 16);
                oacc[f] = wmma_bf16(ap[ks], bv, oacc[f]);
            }
        __syncthreads();
    }

    // normalize and write o[b, row, h*64 + d] as bf16
#pragma unroll
    for (int f = 0; f < 4; ++f)
#pragma unroll
        for (int r = 0; r < 8; ++r) {
            size_t row = (size_t)b * KSEL + q0 + wave * 16 + r + 8 * g;
            o[row * DMODEL + h * HD + f * 16 + lm] = f2bf(oacc[f][r] / lstate[r]);
        }
}

// ---------------------------------------------------------------------------
// Host orchestration
// ---------------------------------------------------------------------------
extern "C" void kernel_launch(void* const* d_in, const int* in_sizes, int n_in,
                              void* d_out, int out_size, void* d_ws, size_t ws_size,
                              hipStream_t stream) {
    const float* x        = (const float*)d_in[0];
    const float* w_router = (const float*)d_in[1];
    const float* ln1_s    = (const float*)d_in[2];
    const float* ln1_b    = (const float*)d_in[3];
    const float* ln2_s    = (const float*)d_in[4];
    const float* ln2_b    = (const float*)d_in[5];
    const float* w_qkv    = (const float*)d_in[6];
    const float* w_out    = (const float*)d_in[7];
    const float* w1       = (const float*)d_in[8];
    const float* w2       = (const float*)d_in[9];
    const float* w3       = (const float*)d_in[10];
    float* out = (float*)d_out;

    const int MR = BATCH * KSEL;                       // 8192 selected rows
    char* ws = (char*)d_ws;
    size_t off = 0;
    auto alloc = [&](size_t bytes) -> char* {
        char* p = ws + off;
        off += (bytes + 255) & ~(size_t)255;
        return p;
    };
    float*          scores  = (float*)alloc((size_t)BATCH * TSEQ * 4);
    int*            idx     = (int*)alloc((size_t)BATCH * KSEL * 4);
    unsigned short* bWqkv   = (unsigned short*)alloc((size_t)3 * DMODEL * DMODEL * 2);
    unsigned short* bWout   = (unsigned short*)alloc((size_t)DMODEL * DMODEL * 2);
    unsigned short* bW1     = (unsigned short*)alloc((size_t)DFFDIM * DMODEL * 2);
    unsigned short* bW2     = (unsigned short*)alloc((size_t)DFFDIM * DMODEL * 2);
    unsigned short* bW3     = (unsigned short*)alloc((size_t)DMODEL * DFFDIM * 2);
    unsigned short* normed  = (unsigned short*)alloc((size_t)MR * DMODEL * 2);   // LN1 then LN2
    unsigned short* qkvbuf  = (unsigned short*)alloc((size_t)MR * 3 * DMODEL * 2);
    unsigned short* obuf    = (unsigned short*)alloc((size_t)MR * DMODEL * 2);
    float*          xr      = (float*)alloc((size_t)MR * DMODEL * 4);
    unsigned short* gbuf    = qkvbuf;  // (qkv 48MB + o 16MB) reused as g (64MB)

    // 0) weights -> bf16
    int nqkv = 3 * DMODEL * DMODEL, nsq = DMODEL * DMODEL, nff = DFFDIM * DMODEL;
    cvt_bf16_kernel<<<(nqkv + 255) / 256, 256, 0, stream>>>(w_qkv, bWqkv, nqkv);
    cvt_bf16_kernel<<<(nsq  + 255) / 256, 256, 0, stream>>>(w_out, bWout, nsq);
    cvt_bf16_kernel<<<(nff  + 255) / 256, 256, 0, stream>>>(w1, bW1, nff);
    cvt_bf16_kernel<<<(nff  + 255) / 256, 256, 0, stream>>>(w2, bW2, nff);
    cvt_bf16_kernel<<<(nff  + 255) / 256, 256, 0, stream>>>(w3, bW3, nff);

    // 1) router + top-k + aux loss
    router_kernel<<<BATCH * TSEQ / 8, 256, 0, stream>>>(x, w_router, scores);
    topk_kernel<<<BATCH, 1024, 0, stream>>>(scores, idx);
    aux_kernel<<<1, 128, 0, stream>>>(scores, out + (size_t)BATCH * TSEQ * DMODEL);

    // 2) pass-through tokens: out = x everywhere (selected rows overwritten later)
    hipMemcpyAsync(out, x, (size_t)BATCH * TSEQ * DMODEL * sizeof(float),
                   hipMemcpyDeviceToDevice, stream);

    // 3) gather + LN1, QKV projection (WMMA + TDM)
    ln_kernel<<<MR / 8, 256, 0, stream>>>(x, idx, ln1_s, ln1_b, normed, 1);
    gemm_kernel<0><<<dim3(3 * DMODEL / 128, MR / 128), 256, 0, stream>>>(
        normed, bWqkv, MR, 3 * DMODEL, DMODEL, qkvbuf, nullptr, nullptr, nullptr);

    // 4) flash attention (WMMA)
    attn_kernel<<<dim3(KSEL / 64, NHEAD, BATCH), 128, 0, stream>>>(qkvbuf, obuf);

    // 5) out-proj + gathered residual -> xr (WMMA + TDM)
    gemm_kernel<1><<<dim3(DMODEL / 128, MR / 128), 256, 0, stream>>>(
        obuf, bWout, MR, DMODEL, DMODEL, xr, x, nullptr, idx);

    // 6) LN2, fused SwiGLU dual-GEMM, final GEMM + residual + scatter (WMMA + TDM)
    ln_kernel<<<MR / 8, 256, 0, stream>>>(xr, nullptr, ln2_s, ln2_b, normed, 0);
    gemm_swiglu_kernel<<<dim3(DFFDIM / 64, MR / 128), 256, 0, stream>>>(
        normed, bW1, bW2, MR, DFFDIM, DMODEL, gbuf);
    gemm_kernel<2><<<dim3(DMODEL / 128, MR / 128), 256, 0, stream>>>(
        gbuf, bW3, MR, DMODEL, DFFDIM, out, nullptr, xr, idx);

    (void)in_sizes; (void)n_in; (void)out_size; (void)ws_size;
}
